// Block_79680233275670
// MI455X (gfx1250) — compile-verified, matrix operand order checked by reading, not run on hardware
//
#include <hip/hip_runtime.h>
#include <hip/hip_bf16.h>
#include <stdint.h>

// ---------------------------------------------------------------------------
// Types for CDNA5 WMMA (gfx1250, wave32)
// ---------------------------------------------------------------------------
typedef __attribute__((ext_vector_type(16))) __bf16 v16bf;
typedef __attribute__((ext_vector_type(8)))  float  v8f;
typedef int gvec4i __attribute__((vector_size(16)));   // matches builtin param

union FragBF { v16bf v; unsigned short u[16]; uint4 q[2]; };
union AccF   { v8f   v; float f[8]; };

// Async global->LDS path (CDNA5): tracked by ASYNCcnt, no VGPR round trip.
#if defined(__has_builtin)
#if __has_builtin(__builtin_amdgcn_global_load_async_to_lds_b128) && \
    __has_builtin(__builtin_amdgcn_s_wait_asynccnt)
#define HAVE_ASYNC_LDS 1
#endif
#endif
#ifndef HAVE_ASYNC_LDS
#define HAVE_ASYNC_LDS 0
#endif

#if HAVE_ASYNC_LDS
__device__ __forceinline__ void async_b128(const unsigned short* g,
                                           unsigned short* l) {
    unsigned short* gnc = const_cast<unsigned short*>(g);
    __builtin_amdgcn_global_load_async_to_lds_b128(
        (__attribute__((address_space(1))) gvec4i*)gnc,
        (__attribute__((address_space(3))) gvec4i*)l,
        /*offset=*/0, /*cpol=*/0);
}
#endif

__device__ __forceinline__ unsigned short f32_to_bf16u(float f) {
    unsigned int u = __float_as_uint(f);
    unsigned int r = (u + 0x7FFFu + ((u >> 16) & 1u)) >> 16;  // RNE
    return (unsigned short)r;
}
__device__ __forceinline__ float bf16u_to_f32(unsigned short h) {
    return __uint_as_float(((unsigned int)h) << 16);
}

__device__ __forceinline__ v8f zero_acc() {
    AccF a;
#pragma unroll
    for (int i = 0; i < 8; ++i) a.f[i] = 0.0f;
    return a.v;
}

// A-fragment (16x32, 16-bit) per ISA layout:
//  lane<16 : elems 0..7 -> K=k0..k0+7,   elems 8..15 -> K=k0+16..k0+23
//  lane>=16: elems 0..7 -> K=k0+8..k0+15, elems 8..15 -> K=k0+24..k0+31
// B-fragment (32x16) is the mirror with lane&15 = N column, source stored
// row-major [n][k] (i.e. column-major B), so both use the same loader.
__device__ __forceinline__ v16bf load_frag(const unsigned short* base,
                                           int rowStride, int k0) {
    const int lane = threadIdx.x & 31;
    const int r    = lane & 15;
    const int kb   = k0 + ((lane >> 4) << 3);
    const unsigned short* p = base + (size_t)r * rowStride + kb;
    FragBF f;
    f.q[0] = *(const uint4*)(p);        // K = kb .. kb+7
    f.q[1] = *(const uint4*)(p + 16);   // K = kb+16 .. kb+23
    return f.v;
}

__device__ __forceinline__ v8f wmma_bf16(v16bf a, v16bf b, v8f c) {
    return __builtin_amdgcn_wmma_f32_16x16x32_bf16(
        /*neg_a=*/false, a, /*neg_b=*/false, b,
        /*c_mod=*/(short)0, c, /*reuse_a=*/false, /*reuse_b=*/false);
}

__device__ __forceinline__ float hswish(float v) {
    float c = fminf(fmaxf(v + 3.0f, 0.0f), 6.0f);
    return v * c * (1.0f / 6.0f);
}

// ---------------------------------------------------------------------------
// f32 -> bf16 weight conversion
// ---------------------------------------------------------------------------
__global__ __launch_bounds__(256) void cvt_bf16_kernel(
    const float* __restrict__ in, unsigned short* __restrict__ out, int n) {
    int i = blockIdx.x * 256 + threadIdx.x;
    if (i < n) out[i] = f32_to_bf16u(in[i]);
}

// ---------------------------------------------------------------------------
// LayerNorm over C=384, one wave per row (wave32), bf16 output
// ---------------------------------------------------------------------------
__global__ __launch_bounds__(256) void layernorm_bf16_kernel(
    const float* __restrict__ x, const float* __restrict__ g,
    const float* __restrict__ bb, unsigned short* __restrict__ out, int M) {
    const int C = 384;
    const int w = threadIdx.x >> 5, lane = threadIdx.x & 31;
    const int row = blockIdx.x * 8 + w;
    if (row >= M) return;
    const float* px = x + (size_t)row * C;
    float vals[12], s = 0.0f;
#pragma unroll
    for (int i = 0; i < 12; ++i) { vals[i] = px[lane + 32 * i]; s += vals[i]; }
#pragma unroll
    for (int off = 16; off; off >>= 1) s += __shfl_xor(s, off, 32);
    const float mu = s * (1.0f / C);
    float vs = 0.0f;
#pragma unroll
    for (int i = 0; i < 12; ++i) { float d = vals[i] - mu; vs += d * d; }
#pragma unroll
    for (int off = 16; off; off >>= 1) vs += __shfl_xor(vs, off, 32);
    const float inv = rsqrtf(vs * (1.0f / C) + 1e-5f);
    unsigned short* po = out + (size_t)row * C;
#pragma unroll
    for (int i = 0; i < 12; ++i) {
        int c = lane + 32 * i;
        po[c] = f32_to_bf16u((vals[i] - mu) * inv * g[c] + bb[c]);
    }
}

// ---------------------------------------------------------------------------
// Generic bf16 WMMA GEMM: out[m][n] = act(sum_k A[m][k]*W[n][k] + bias[n]) + res
// Block tile 128x128, 8 waves as 4x2, each wave 32x64 (2x4 WMMA tiles).
// Double-buffered LDS fed by async global->LDS (ASYNCcnt), 1 barrier / K-step.
// flags: bit0 = bf16 output, bit1 = hardswish
// ---------------------------------------------------------------------------
#define GEMM_LDA 40   // padded LDS row stride (elems); 80B keeps 16B alignment

__global__ __launch_bounds__(256) void gemm_bf16_kernel(
    const unsigned short* __restrict__ A, const unsigned short* __restrict__ W,
    const float* __restrict__ bias, const float* __restrict__ residual,
    void* __restrict__ out, int M, int N, int K, int flags) {
    __shared__ unsigned short As[2][128 * GEMM_LDA];
    __shared__ unsigned short Ws[2][128 * GEMM_LDA];
    const int t = threadIdx.x;
    const int w = t >> 5, lane = t & 31;
    const int bm = blockIdx.y * 128;
    const int bn = blockIdx.x * 128;
    const int wm = (w >> 1) * 32;  // 0,32,64,96
    const int wn = (w & 1) * 64;   // 0,64

    // staging assignment: thread -> (row, 16-elem column half)
    const int srow = t >> 1, sch = (t & 1) * 16;
    const unsigned short* gA = A + (size_t)(bm + srow) * K + sch;
    const unsigned short* gW = W + (size_t)(bn + srow) * K + sch;
    const int lOff = srow * GEMM_LDA + sch;

    v8f acc[2][4];
#pragma unroll
    for (int i = 0; i < 2; ++i)
#pragma unroll
        for (int j = 0; j < 4; ++j) acc[i][j] = zero_acc();

    const int KT = K >> 5;  // K / 32

    auto stage = [&](int kt, int buf) {
        const unsigned short* a  = gA + kt * 32;
        const unsigned short* ww = gW + kt * 32;
#if HAVE_ASYNC_LDS
        async_b128(a,      &As[buf][lOff]);
        async_b128(a + 8,  &As[buf][lOff + 8]);
        async_b128(ww,     &Ws[buf][lOff]);
        async_b128(ww + 8, &Ws[buf][lOff + 8]);
#else
        uint4 a0 = *(const uint4*)(a);
        uint4 a1 = *(const uint4*)(a + 8);
        uint4 w0 = *(const uint4*)(ww);
        uint4 w1 = *(const uint4*)(ww + 8);
        *(uint4*)(&As[buf][lOff])     = a0;
        *(uint4*)(&As[buf][lOff + 8]) = a1;
        *(uint4*)(&Ws[buf][lOff])     = w0;
        *(uint4*)(&Ws[buf][lOff + 8]) = w1;
#endif
    };

    auto compute = [&](int buf) {
        v16bf a0 = load_frag(&As[buf][(wm)      * GEMM_LDA], GEMM_LDA, 0);
        v16bf a1 = load_frag(&As[buf][(wm + 16) * GEMM_LDA], GEMM_LDA, 0);
        v16bf bfr[4];
#pragma unroll
        for (int j = 0; j < 4; ++j)
            bfr[j] = load_frag(&Ws[buf][(wn + 16 * j) * GEMM_LDA], GEMM_LDA, 0);
#pragma unroll
        for (int j = 0; j < 4; ++j) {
            acc[0][j] = wmma_bf16(a0, bfr[j], acc[0][j]);
            acc[1][j] = wmma_bf16(a1, bfr[j], acc[1][j]);
        }
    };

#if HAVE_ASYNC_LDS
    stage(0, 0);
    for (int kt = 0; kt < KT; ++kt) {
        __builtin_amdgcn_s_wait_asynccnt(0);  // this wave's tile writes done
        __syncthreads();                      // everyone's tile writes visible
        if (kt + 1 < KT) stage(kt + 1, (kt + 1) & 1);  // overlaps WMMAs below
        compute(kt & 1);
    }
#else
    for (int kt = 0; kt < KT; ++kt) {
        stage(kt, 0);
        __syncthreads();
        compute(0);
        __syncthreads();
    }
#endif

    // Epilogue.  D layout: elem r -> row r (lanes 0-15) / row r+8 (lanes 16-31)
    const int hi8 = (lane >> 4) << 3;
    const int cn  = lane & 15;
#pragma unroll
    for (int ti = 0; ti < 2; ++ti)
#pragma unroll
        for (int tj = 0; tj < 4; ++tj) {
            AccF ac; ac.v = acc[ti][tj];
            const int ncol = bn + wn + tj * 16 + cn;
            const float bv = bias ? bias[ncol] : 0.0f;
#pragma unroll
            for (int r = 0; r < 8; ++r) {
                const int mrow = bm + wm + ti * 16 + hi8 + r;
                float v = ac.f[r] + bv;
                if (flags & 2) v = hswish(v);
                const size_t oi = (size_t)mrow * N + ncol;
                if (residual) v += residual[oi];
                if (flags & 1) ((unsigned short*)out)[oi] = f32_to_bf16u(v);
                else           ((float*)out)[oi] = v;
            }
        }
}

// ---------------------------------------------------------------------------
// V transpose: qkv[b,n,1152] (V slice) -> vt[(b*4+h)*96+d][n]  (bf16)
// ---------------------------------------------------------------------------
__global__ __launch_bounds__(256) void vtrans_kernel(
    const unsigned short* __restrict__ qkv, unsigned short* __restrict__ vt) {
    int idx = blockIdx.x * 256 + threadIdx.x;       // 16*4*96*1024
    int n  = idx & 1023;
    int dd = (idx >> 10) % 96;
    int bh = (idx >> 10) / 96;
    int b = bh >> 2, h = bh & 3;
    vt[idx] = qkv[((size_t)(b * 1024 + n)) * 1152 + h * 288 + 192 + dd];
}

// ---------------------------------------------------------------------------
// Fused attention: one block per (b, h, 32-row tile).
// LDS: Q 32x96 bf16 | S 32x1024 f32 | P 32x1024 bf16  (~198 KB of 320 KB/WGP)
// ---------------------------------------------------------------------------
__global__ __launch_bounds__(256) void attn_kernel(
    const unsigned short* __restrict__ qkv, const unsigned short* __restrict__ vt,
    unsigned short* __restrict__ ob) {
    extern __shared__ char smem[];
    unsigned short* Qs = (unsigned short*)smem;                       // 6144 B
    float*          Ss = (float*)(smem + 32 * 96 * 2);                // 131072 B
    unsigned short* Ps = (unsigned short*)(smem + 32 * 96 * 2 + 32 * 1024 * 4);

    const int t = threadIdx.x, w = t >> 5, lane = t & 31;
    const int gid = blockIdx.x;
    const int rb = gid & 31;          // row block (32 rows each)
    const int h  = (gid >> 5) & 3;
    const int b  = gid >> 7;

    // ---- stage Q rows into LDS (coalesced 32-bit loads) ----
#pragma unroll
    for (int i = 0; i < 6; ++i) {
        int idx = t + 256 * i;                // < 1536 uints
        int e = idx * 2;
        int row = e / 96, d = e % 96;
        unsigned int vv = *(const unsigned int*)(
            qkv + ((size_t)(b * 1024 + rb * 32 + row)) * 1152 + h * 288 + d);
        *(unsigned int*)(Qs + e) = vv;
    }
    __syncthreads();

    // ---- phase 1: S = (Q K^T) * scale  via WMMA ----
    {
        const int rt = (w & 1) * 16;          // row tile within block
        const int ctbase = (w >> 1) * 16;     // 4 wave-columns x 16 tiles
        const float scale = 0.10206207261596575f;  // 1/sqrt(96)
        const int hi8 = (lane >> 4) << 3, cn = lane & 15;
        for (int tt = 0; tt < 16; ++tt) {
            const int ct = ctbase + tt;       // col tile 0..63
            v8f acc = zero_acc();
            const unsigned short* kbaseP =
                qkv + ((size_t)(b * 1024 + ct * 16)) * 1152 + h * 288 + 96;
#pragma unroll
            for (int d0 = 0; d0 < 96; d0 += 32) {
                v16bf af = load_frag(Qs + rt * 96, 96, d0);
                v16bf bf = load_frag(kbaseP, 1152, d0);
                acc = wmma_bf16(af, bf, acc);
            }
            AccF ac; ac.v = acc;
#pragma unroll
            for (int r = 0; r < 8; ++r)
                Ss[(rt + hi8 + r) * 1024 + ct * 16 + cn] = ac.f[r] * scale;
        }
    }
    __syncthreads();

    // ---- phase 2: row softmax with wave32 shuffle reductions ----
#pragma unroll
    for (int i = 0; i < 4; ++i) {
        const int row = w * 4 + i;
        float* pr = Ss + row * 1024;
        float m = -3.0e38f;
        for (int j = lane; j < 1024; j += 32) m = fmaxf(m, pr[j]);
#pragma unroll
        for (int off = 16; off; off >>= 1) m = fmaxf(m, __shfl_xor(m, off, 32));
        float s = 0.0f;
        for (int j = lane; j < 1024; j += 32) {
            float e = __expf(pr[j] - m);
            pr[j] = e; s += e;
        }
#pragma unroll
        for (int off = 16; off; off >>= 1) s += __shfl_xor(s, off, 32);
        const float inv = 1.0f / s;
        unsigned short* pp = Ps + row * 1024;
        for (int j = lane; j < 1024; j += 32) pp[j] = f32_to_bf16u(pr[j] * inv);
    }
    __syncthreads();

    // ---- phase 3: O = P @ V via WMMA (V pre-transposed) ----
    for (int tid = w; tid < 12; tid += 8) {   // 2 row tiles x 6 col tiles
        const int rt  = (tid / 6) * 16;
        const int ctd = tid % 6;
        v8f acc = zero_acc();
        const unsigned short* vbase =
            vt + ((size_t)((b * 4 + h) * 96 + ctd * 16)) * 1024;
        for (int k0 = 0; k0 < 1024; k0 += 32) {
            v16bf af = load_frag(Ps + rt * 1024, 1024, k0);
            v16bf bf = load_frag(vbase, 1024, k0);
            acc = wmma_bf16(af, bf, acc);
        }
        AccF ac; ac.v = acc;
        const int hi8 = (lane >> 4) << 3, cn = lane & 15;
#pragma unroll
        for (int r = 0; r < 8; ++r) {
            const int row = rb * 32 + rt + hi8 + r;
            const int d   = ctd * 16 + cn;
            ob[((size_t)(b * 1024 + row)) * 384 + h * 96 + d] =
                f32_to_bf16u(ac.f[r]);
        }
    }
}

// ---------------------------------------------------------------------------
// Depthwise 3x3 conv (SAME, zero pad) + hardswish, bf16 in/out.
// h1/h2 layout: [b*1024 + y*32 + x][1536]  -> channel dim is coalesced.
// ---------------------------------------------------------------------------
__global__ __launch_bounds__(256) void dwconv_kernel(
    const unsigned short* __restrict__ h1, const float* __restrict__ Wdw,
    const float* __restrict__ bdw, unsigned short* __restrict__ h2) {
    const int CH = 1536;
    size_t idx = (size_t)blockIdx.x * 256 + threadIdx.x;  // 16*1024*1536
    const int j = (int)(idx % CH);
    const size_t m = idx / CH;              // b*1024 + y*32 + x
    const int xx = (int)(m & 31);
    const int yy = (int)((m >> 5) & 31);
    const size_t bbase = (m >> 10) << 10;   // b*1024
    float acc = bdw[j];
#pragma unroll
    for (int ky = 0; ky < 3; ++ky) {
        const int y2 = yy + ky - 1;
        if ((unsigned)y2 >= 32u) continue;
#pragma unroll
        for (int kx = 0; kx < 3; ++kx) {
            const int x2 = xx + kx - 1;
            if ((unsigned)x2 >= 32u) continue;
            acc += Wdw[j * 9 + ky * 3 + kx] *
                   bf16u_to_f32(h1[(bbase + (size_t)y2 * 32 + x2) * CH + j]);
        }
    }
    h2[idx] = f32_to_bf16u(hswish(acc));
}

// ---------------------------------------------------------------------------
// Host launcher
// ---------------------------------------------------------------------------
extern "C" void kernel_launch(void* const* d_in, const int* in_sizes, int n_in,
                              void* d_out, int out_size, void* d_ws, size_t ws_size,
                              hipStream_t stream) {
    (void)in_sizes; (void)n_in; (void)out_size; (void)ws_size;
    const int M = 16 * 1024;     // B*N rows
    const int C = 384, H3C = 1152, HID = 1536;

    const float* x      = (const float*)d_in[0];
    const float* ln1_g  = (const float*)d_in[1];
    const float* ln1_b  = (const float*)d_in[2];
    const float* Wqkv   = (const float*)d_in[3];
    const float* Wproj  = (const float*)d_in[4];
    const float* bproj  = (const float*)d_in[5];
    const float* ln2_g  = (const float*)d_in[6];
    const float* ln2_b  = (const float*)d_in[7];
    const float* Wfc1   = (const float*)d_in[8];
    const float* bfc1   = (const float*)d_in[9];
    const float* Wdw    = (const float*)d_in[10];
    const float* bdw    = (const float*)d_in[11];
    const float* Wfc2   = (const float*)d_in[12];
    const float* bfc2   = (const float*)d_in[13];

    // workspace layout (256B aligned), with reuse
    char* ws = (char*)d_ws;
    size_t off = 0;
    auto alloc = [&](size_t bytes) {
        size_t o = off;
        off = (off + bytes + 255) & ~(size_t)255;
        return o;
    };
    const size_t oWQKV  = alloc((size_t)H3C * C * 2);
    const size_t oWPROJ = alloc((size_t)C * C * 2);
    const size_t oWFC1  = alloc((size_t)HID * C * 2);
    const size_t oWFC2  = alloc((size_t)C * HID * 2);
    const size_t oXRES  = alloc((size_t)M * C * 4);          // f32, persists
    const size_t oXN    = alloc((size_t)M * C * 2);          // reused as LN2 out
    const size_t oQKV   = alloc((size_t)M * H3C * 2);        // reused as h1 ...
    const size_t oVT    = alloc((size_t)64 * 96 * 1024 * 2); // ... together
    const size_t oOB    = alloc((size_t)M * C * 2);
    const size_t oH2    = alloc((size_t)M * HID * 2);

    unsigned short* wqkv_bf  = (unsigned short*)(ws + oWQKV);
    unsigned short* wproj_bf = (unsigned short*)(ws + oWPROJ);
    unsigned short* wfc1_bf  = (unsigned short*)(ws + oWFC1);
    unsigned short* wfc2_bf  = (unsigned short*)(ws + oWFC2);
    float*          xres     = (float*)(ws + oXRES);
    unsigned short* xn_bf    = (unsigned short*)(ws + oXN);
    unsigned short* qkv_bf   = (unsigned short*)(ws + oQKV);
    unsigned short* vt_bf    = (unsigned short*)(ws + oVT);
    unsigned short* ob_bf    = (unsigned short*)(ws + oOB);
    unsigned short* h1_bf    = qkv_bf;   // 37.75MB + 12.58MB == M*1536*2 exactly
    unsigned short* h2_bf    = (unsigned short*)(ws + oH2);

    // 1) weights -> bf16
    cvt_bf16_kernel<<<(H3C * C + 255) / 256, 256, 0, stream>>>(Wqkv,  wqkv_bf,  H3C * C);
    cvt_bf16_kernel<<<(C * C   + 255) / 256, 256, 0, stream>>>(Wproj, wproj_bf, C * C);
    cvt_bf16_kernel<<<(HID * C + 255) / 256, 256, 0, stream>>>(Wfc1,  wfc1_bf,  HID * C);
    cvt_bf16_kernel<<<(C * HID + 255) / 256, 256, 0, stream>>>(Wfc2,  wfc2_bf,  C * HID);

    // 2) LN1
    layernorm_bf16_kernel<<<M / 8, 256, 0, stream>>>(x, ln1_g, ln1_b, xn_bf, M);

    // 3) QKV GEMM (bf16 out)
    gemm_bf16_kernel<<<dim3(H3C / 128, M / 128), 256, 0, stream>>>(
        xn_bf, wqkv_bf, nullptr, nullptr, qkv_bf, M, H3C, C, /*flags=*/1);

    // 4) V transpose for PV B-fragments
    vtrans_kernel<<<(64 * 96 * 1024) / 256, 256, 0, stream>>>(qkv_bf, vt_bf);

    // 5) fused attention (dyn LDS: Q 6144 + S 131072 + P 65536 = 202752 B)
    attn_kernel<<<16 * 4 * 32, 256, 202752, stream>>>(qkv_bf, vt_bf, ob_bf);

    // 6) proj GEMM + bias + residual(x) -> xres (f32)
    gemm_bf16_kernel<<<dim3(C / 128, M / 128), 256, 0, stream>>>(
        ob_bf, wproj_bf, bproj, x, xres, M, C, C, /*flags=*/0);

    // 7) LN2 -> y (reuses xn buffer)
    layernorm_bf16_kernel<<<M / 8, 256, 0, stream>>>(xres, ln2_g, ln2_b, xn_bf, M);

    // 8) fc1 GEMM + bias + hardswish (bf16 out, overlays dead qkv+vt region)
    gemm_bf16_kernel<<<dim3(HID / 128, M / 128), 256, 0, stream>>>(
        xn_bf, wfc1_bf, bfc1, nullptr, h1_bf, M, HID, C, /*flags=*/1 | 2);

    // 9) depthwise 3x3 + hardswish
    dwconv_kernel<<<(unsigned)(((size_t)M * HID) / 256), 256, 0, stream>>>(
        h1_bf, Wdw, bdw, h2_bf);

    // 10) fc2 GEMM + bias + residual(xres) -> d_out (f32)
    gemm_bf16_kernel<<<dim3(C / 128, M / 128), 256, 0, stream>>>(
        h2_bf, wfc2_bf, bfc2, xres, (float*)d_out, M, C, HID, /*flags=*/0);
}
